// SymplecticHamiltonianDynamic_28346784153888
// MI455X (gfx1250) — matrix operand order repaired and checked by reading, not compiled, not run on hardware
//
#include <hip/hip_runtime.h>
#include <hip/hip_fp16.h>

typedef __attribute__((ext_vector_type(16))) _Float16 v16h;
typedef __attribute__((ext_vector_type(8)))  float    v8f;

static constexpr int Bn = 2, Sn = 2048, Dn = 768, Hn = 12, DKn = 64;
static constexpr int Mrows = Bn * Sn;  // 4096
static constexpr float LN_EPS = 1e-5f;

union HFrag { v16h v; unsigned int u[8]; };

// 16-byte async global->LDS copy (CDNA5 async DMA path), sync fallback.
__device__ __forceinline__ void async_copy16(void* ldsDst, const void* gSrc) {
#if defined(__gfx1250__)
  unsigned lo = (unsigned)(size_t)ldsDst;
  asm volatile("global_load_async_to_lds_b128 %0, %1, off" ::"v"(lo), "v"(gSrc)
               : "memory");
#else
  *(uint4*)ldsDst = *(const uint4*)gSrc;
#endif
}
__device__ __forceinline__ void async_wait() {
#if defined(__gfx1250__)
  asm volatile("s_wait_asynccnt 0" ::: "memory");
#endif
}

// ---------------------------------------------------------------------------
// Prep: x (f32) -> xH (f16), 8 elements/thread
// ---------------------------------------------------------------------------
__global__ __launch_bounds__(256) void x_to_h(const float* __restrict__ x,
                                              _Float16* __restrict__ xh) {
  long i = ((long)blockIdx.x * 256 + threadIdx.x) * 8;
  float4 a = *(const float4*)&x[i];
  float4 b = *(const float4*)&x[i + 4];
  _Float16 h[8] = {(_Float16)a.x, (_Float16)a.y, (_Float16)a.z, (_Float16)a.w,
                   (_Float16)b.x, (_Float16)b.y, (_Float16)b.z, (_Float16)b.w};
  *(uint4*)&xh[i] = *(uint4*)h;
}

// ---------------------------------------------------------------------------
// Prep: transpose + convert the 5 weight matrices to f16 [n][k].
// Grid (24,24,5), 256 threads, 32x32 tiles through LDS.
// ---------------------------------------------------------------------------
__global__ __launch_bounds__(256) void w_transpose(
    const float* __restrict__ W0, const float* __restrict__ W1,
    const float* __restrict__ W2, const float* __restrict__ W3,
    const float* __restrict__ W4, _Float16* __restrict__ dst) {
  const float* W = W0;
  switch (blockIdx.z) {
    case 1: W = W1; break;
    case 2: W = W2; break;
    case 3: W = W3; break;
    case 4: W = W4; break;
    default: break;
  }
  _Float16* out = dst + (size_t)blockIdx.z * Dn * Dn;
  __shared__ _Float16 T[32][32];  // T[n][k]
  const int tid = threadIdx.x;
  const int k0 = blockIdx.x * 32, n0 = blockIdx.y * 32;
  {
    int r = tid >> 3, c4 = (tid & 7) << 2;
    float4 f = *(const float4*)&W[(long)(k0 + r) * Dn + n0 + c4];
    T[c4 + 0][r] = (_Float16)f.x;
    T[c4 + 1][r] = (_Float16)f.y;
    T[c4 + 2][r] = (_Float16)f.z;
    T[c4 + 3][r] = (_Float16)f.w;
  }
  __syncthreads();
  if (tid < 128) {
    int c = tid >> 2, rc = (tid & 3) << 3;
    *(uint4*)&out[(long)(n0 + c) * Dn + k0 + rc] = *(uint4*)&T[c][rc];
  }
}

// ---------------------------------------------------------------------------
// GEMM: C[M,N] = A[M,K] * W[K,N]; A f16 row-major, W pre-transposed f16 [n][k].
// Block tile 128x128, 8 waves, each 32x64 (2x4 WMMA tiles). Double-buffered
// async LDS DMA staging (DMA for tile i+1 hidden behind WMMA of tile i).
// Epilogue (runtime-selected): flat f32 +bias+resid, or heads layouts
// (f32 / f16*hscale / transposed f16 [B,H,DK,S]).
// ---------------------------------------------------------------------------
__global__ __launch_bounds__(256) void gemm128x128(
    const _Float16* __restrict__ Ah, const _Float16* __restrict__ Wt,
    float* __restrict__ oF32Heads, _Float16* __restrict__ oF16Heads,
    _Float16* __restrict__ oVT, float* __restrict__ oFlat,
    const float* __restrict__ bias, const float* __restrict__ resid,
    float hscale) {
  constexpr int Kdim = Dn, Ndim = Dn;
  __shared__ _Float16 sA[2][128][32];
  __shared__ _Float16 sBt[2][128][32];

  const int tid  = threadIdx.x;
  const int lane = tid & 31;
  const int wave = tid >> 5;
  const int half = lane >> 4;
  const int l16  = lane & 15;
  const int wm   = wave & 3;
  const int wn   = wave >> 2;
  const int m0   = blockIdx.y * 128;
  const int n0   = blockIdx.x * 128;

  auto stage = [&](int buf, int k0) {
#pragma unroll
    for (int e = 0; e < 2; ++e) {  // A: 512 16B chunks
      int ee = tid + e * 256;
      int r = ee >> 2, kc = (ee & 3) << 3;
      async_copy16(&sA[buf][r][kc], Ah + (long)(m0 + r) * Kdim + k0 + kc);
    }
#pragma unroll
    for (int e = 0; e < 2; ++e) {  // Wt: 512 16B chunks
      int ee = tid + e * 256;
      int n = ee >> 2, kc = (ee & 3) << 3;
      async_copy16(&sBt[buf][n][kc], Wt + (long)(n0 + n) * Kdim + k0 + kc);
    }
  };

  v8f acc[2][4];
  for (int a = 0; a < 2; ++a)
    for (int c = 0; c < 4; ++c)
      for (int i = 0; i < 8; ++i) acc[a][c][i] = 0.0f;

  stage(0, 0);
  constexpr int nIter = Kdim / 32;
  for (int it = 0; it < nIter; ++it) {
    const int cur = it & 1;
    async_wait();      // my DMAs for tile `it` done
    __syncthreads();   // everyone's DMAs done; prev reads of other buf done
    if (it + 1 < nIter) stage(cur ^ 1, (it + 1) * 32);

    HFrag a[2], bf[4];
#pragma unroll
    for (int rt = 0; rt < 2; ++rt) {
      int m = wm * 32 + rt * 16 + l16;
#pragma unroll
      for (int i = 0; i < 8; ++i) {
        int kb = ((i & 4) << 2) + half * 8 + ((i & 3) << 1);
        a[rt].u[i] = *(const unsigned int*)&sA[cur][m][kb];
      }
    }
#pragma unroll
    for (int ct = 0; ct < 4; ++ct) {
      int n = wn * 64 + ct * 16 + l16;
#pragma unroll
      for (int i = 0; i < 8; ++i) {
        int kb = ((i & 4) << 2) + half * 8 + ((i & 3) << 1);
        bf[ct].u[i] = *(const unsigned int*)&sBt[cur][n][kb];
      }
    }
#pragma unroll
    for (int rt = 0; rt < 2; ++rt)
#pragma unroll
      for (int ct = 0; ct < 4; ++ct)
        acc[rt][ct] = __builtin_amdgcn_wmma_f32_16x16x32_f16(
            false, a[rt].v, false, bf[ct].v, (short)0, acc[rt][ct], false,
            false);
  }

  for (int rt = 0; rt < 2; ++rt)
    for (int ct = 0; ct < 4; ++ct) {
      int col  = n0 + wn * 64 + ct * 16 + l16;
      int row0 = m0 + wm * 32 + rt * 16 + half * 8;
      if (oFlat) {
        for (int i = 0; i < 8; ++i) {
          int row = row0 + i;
          oFlat[(long)row * Ndim + col] =
              acc[rt][ct][i] + bias[col] + resid[(long)row * Ndim + col];
        }
      } else {
        int bi = row0 >> 11, s0 = row0 & (Sn - 1);
        int h = col >> 6, dk = col & 63;
        long hbase = (((long)(bi * Hn + h)) * Sn + s0) * DKn + dk;
        if (oF32Heads)
          for (int i = 0; i < 8; ++i)
            oF32Heads[hbase + (long)i * DKn] = acc[rt][ct][i];
        if (oF16Heads)
          for (int i = 0; i < 8; ++i)
            oF16Heads[hbase + (long)i * DKn] =
                (_Float16)(acc[rt][ct][i] * hscale);
        if (oVT) {  // [B,H,DK,S]; 8 consecutive s -> one 16B store
          _Float16 h8[8];
          for (int i = 0; i < 8; ++i) h8[i] = (_Float16)acc[rt][ct][i];
          *(uint4*)&oVT[((long)(bi * Hn + h) * DKn + dk) * Sn + s0] =
              *(const uint4*)h8;
        }
      }
    }
}

// ---------------------------------------------------------------------------
// Fused single-pass attention + leapfrog. 128 threads (4 waves), one
// 32-query tile of one (b,h); 64-key tiles, double-buffered async staging,
// shfl-based online softmax (3 barriers/iter). q pre-scaled by 1/sqrt(DK).
// ---------------------------------------------------------------------------
__global__ __launch_bounds__(128) void attn_leapfrog(
    const _Float16* __restrict__ qH, const _Float16* __restrict__ kH,
    const _Float16* __restrict__ vT, const float* __restrict__ qF,
    const float* __restrict__ pF, const float* __restrict__ dtp,
    const float* __restrict__ gatep, _Float16* __restrict__ qnewH) {
  const int qt = blockIdx.x & 63;
  const int bh = blockIdx.x >> 6;
  const int bi = bh / Hn, h = bh % Hn;

  __shared__ _Float16 sK[2][64][64];   // K tile [t][d]
  __shared__ _Float16 sVt[2][64][64];  // V^T tile [d][t]
  __shared__ float    sE[32][64];
  __shared__ _Float16 sP[32][64];
  __shared__ float sMax[32], sSum[32], sAlpha[32];

  const int tid  = threadIdx.x;
  const int lane = tid & 31;
  const int wave = tid >> 5;
  const int half = lane >> 4;
  const int l16  = lane & 15;
  const int mt   = wave & 1;
  const int et   = wave >> 1;
  const int srow = tid >> 2, spart = tid & 3;

  const float dt   = dtp[0];
  const float gate = gatep[0];

  const _Float16* kAll  = kH + (long)bh * Sn * DKn;
  const _Float16* vTAll = vT + (long)bh * DKn * Sn;

  auto stageKV = [&](int buf, int kt) {
#pragma unroll
    for (int e = 0; e < 4; ++e) {  // K: 512 16B chunks
      int ee = tid + e * 128;
      int t = ee >> 3, dc = (ee & 7) << 3;
      async_copy16(&sK[buf][t][dc], kAll + (long)(kt * 64 + t) * DKn + dc);
    }
#pragma unroll
    for (int e = 0; e < 4; ++e) {  // V^T: 512 16B chunks
      int ee = tid + e * 128;
      int d = ee >> 3, tc = (ee & 7) << 3;
      async_copy16(&sVt[buf][d][tc], vTAll + (long)d * Sn + kt * 64 + tc);
    }
  };

  HFrag qa[2];
  {
    int s = qt * 32 + mt * 16 + l16;
    const _Float16* qrow = qH + ((long)bh * Sn + s) * DKn;
#pragma unroll
    for (int f = 0; f < 2; ++f)
#pragma unroll
      for (int i = 0; i < 8; ++i) {
        int kb = f * 32 + ((i & 4) << 2) + half * 8 + ((i & 3) << 1);
        qa[f].u[i] = *(const unsigned int*)&qrow[kb];
      }
  }

  v8f o[2];
  for (int c = 0; c < 2; ++c)
    for (int i = 0; i < 8; ++i) o[c][i] = 0.0f;
  if (tid < 32) { sMax[tid] = -1e30f; sSum[tid] = 0.0f; }

  stageKV(0, 0);
  constexpr int nIter = Sn / 64;
  for (int kt = 0; kt < nIter; ++kt) {
    const int cur = kt & 1;
    async_wait();
    __syncthreads();
    if (kt + 1 < nIter) stageKV(cur ^ 1, kt + 1);

    // e = q k^T : wave computes rows mt, key cols [et*32, et*32+32)
#pragma unroll
    for (int nt = 0; nt < 2; ++nt) {
      v8f ec;
      for (int i = 0; i < 8; ++i) ec[i] = 0.0f;
      HFrag kfr[2];
      int t = (et * 2 + nt) * 16 + l16;
#pragma unroll
      for (int f = 0; f < 2; ++f)
#pragma unroll
        for (int i = 0; i < 8; ++i) {
          int kb = f * 32 + ((i & 4) << 2) + half * 8 + ((i & 3) << 1);
          kfr[f].u[i] = *(const unsigned int*)&sK[cur][t][kb];
        }
      ec = __builtin_amdgcn_wmma_f32_16x16x32_f16(
          false, qa[0].v, false, kfr[0].v, (short)0, ec, false, false);
      ec = __builtin_amdgcn_wmma_f32_16x16x32_f16(
          false, qa[1].v, false, kfr[1].v, (short)0, ec, false, false);
      for (int i = 0; i < 8; ++i)
        sE[mt * 16 + half * 8 + i][(et * 2 + nt) * 16 + l16] = ec[i];
    }
    __syncthreads();

    // online softmax: 4 consecutive lanes per row, shfl_xor reductions
    {
      float mx = sMax[srow];
      float pmax = -1e30f;
      for (int j = 0; j < 16; ++j)
        pmax = fmaxf(pmax, sE[srow][spart * 16 + j]);
      pmax = fmaxf(pmax, __shfl_xor(pmax, 1, 32));
      pmax = fmaxf(pmax, __shfl_xor(pmax, 2, 32));
      float nmax  = fmaxf(mx, pmax);
      float alpha = __expf(mx - nmax);
      float ps = 0.0f;
      for (int j = 0; j < 16; ++j) {
        float pv = __expf(sE[srow][spart * 16 + j] - nmax);
        sP[srow][spart * 16 + j] = (_Float16)pv;
        ps += pv;
      }
      ps += __shfl_xor(ps, 1, 32);
      ps += __shfl_xor(ps, 2, 32);
      if (spart == 0) {
        sSum[srow]   = sSum[srow] * alpha + ps;
        sMax[srow]   = nmax;
        sAlpha[srow] = alpha;
      }
    }
    __syncthreads();

    // O = alpha*O + P·V  (wave: rows mt, d cols [et*32, et*32+32))
    HFrag pa[2];
    {
      int m = mt * 16 + l16;
#pragma unroll
      for (int f = 0; f < 2; ++f)
#pragma unroll
        for (int i = 0; i < 8; ++i) {
          int kb = f * 32 + ((i & 4) << 2) + half * 8 + ((i & 3) << 1);
          pa[f].u[i] = *(const unsigned int*)&sP[m][kb];
        }
    }
#pragma unroll
    for (int ct = 0; ct < 2; ++ct) {
      HFrag vfr[2];
      int d = et * 32 + ct * 16 + l16;
#pragma unroll
      for (int f = 0; f < 2; ++f)
#pragma unroll
        for (int i = 0; i < 8; ++i) {
          int kb = f * 32 + ((i & 4) << 2) + half * 8 + ((i & 3) << 1);
          vfr[f].u[i] = *(const unsigned int*)&sVt[cur][d][kb];
        }
      for (int i = 0; i < 8; ++i) o[ct][i] *= sAlpha[mt * 16 + half * 8 + i];
      o[ct] = __builtin_amdgcn_wmma_f32_16x16x32_f16(
          false, pa[0].v, false, vfr[0].v, (short)0, o[ct], false, false);
      o[ct] = __builtin_amdgcn_wmma_f32_16x16x32_f16(
          false, pa[1].v, false, vfr[1].v, (short)0, o[ct], false, false);
    }
  }
  __syncthreads();

  const float c2 = 0.5f * dt * dt * gate;
  for (int ct = 0; ct < 2; ++ct)
    for (int i = 0; i < 8; ++i) {
      int rloc = mt * 16 + half * 8 + i;
      int s    = qt * 32 + rloc;
      int d    = et * 32 + ct * 16 + l16;
      float pe = o[ct][i] / sSum[rloc];
      long hoff = ((long)bh * Sn + s) * DKn + d;
      float qn = qF[hoff] + dt * pF[hoff] - c2 * pe;
      qnewH[((long)bi * Sn + s) * Dn + h * DKn + d] = (_Float16)qn;
    }
}

// ---------------------------------------------------------------------------
// Row LayerNorm over D=768 (float4 vectorized)
// ---------------------------------------------------------------------------
__global__ __launch_bounds__(256) void ln_rows(const float* __restrict__ of,
                                               const float* __restrict__ gamma,
                                               const float* __restrict__ beta,
                                               float* __restrict__ out) {
  __shared__ float red[256];
  const int row = blockIdx.x;
  const int tid = threadIdx.x;
  const float4* r4 = (const float4*)(of + (long)row * Dn);
  float4 myv = make_float4(0.f, 0.f, 0.f, 0.f);
  float s = 0.0f;
  if (tid < 192) {
    myv = r4[tid];
    s   = myv.x + myv.y + myv.z + myv.w;
  }
  red[tid] = s;
  __syncthreads();
  for (int st = 128; st > 0; st >>= 1) {
    if (tid < st) red[tid] += red[tid + st];
    __syncthreads();
  }
  float mu = red[0] / Dn;
  __syncthreads();
  float vs = 0.0f;
  if (tid < 192) {
    float a = myv.x - mu, b = myv.y - mu, c = myv.z - mu, d = myv.w - mu;
    vs = a * a + b * b + c * c + d * d;
  }
  red[tid] = vs;
  __syncthreads();
  for (int st = 128; st > 0; st >>= 1) {
    if (tid < st) red[tid] += red[tid + st];
    __syncthreads();
  }
  float rstd = rsqrtf(red[0] / Dn + LN_EPS);
  if (tid < 192) {
    float4 g = ((const float4*)gamma)[tid];
    float4 b = ((const float4*)beta)[tid];
    float4 o;
    o.x = (myv.x - mu) * rstd * g.x + b.x;
    o.y = (myv.y - mu) * rstd * g.y + b.y;
    o.z = (myv.z - mu) * rstd * g.z + b.z;
    o.w = (myv.w - mu) * rstd * g.w + b.w;
    ((float4*)(out + (long)row * Dn))[tid] = o;
  }
}

// ---------------------------------------------------------------------------
extern "C" void kernel_launch(void* const* d_in, const int* in_sizes, int n_in,
                              void* d_out, int out_size, void* d_ws,
                              size_t ws_size, hipStream_t stream) {
  const float* x     = (const float*)d_in[0];
  const float* Wq    = (const float*)d_in[1];
  const float* Wp    = (const float*)d_in[2];
  const float* Wk    = (const float*)d_in[3];
  const float* Wv    = (const float*)d_in[4];
  const float* Wo    = (const float*)d_in[5];
  const float* bo    = (const float*)d_in[6];
  const float* gamma = (const float*)d_in[7];
  const float* beta  = (const float*)d_in[8];
  const float* dtp   = (const float*)d_in[9];
  const float* gatep = (const float*)d_in[10];

  char* ws = (char*)d_ws;
  size_t off = 0;
  auto alloc = [&](size_t bytes) -> void* {
    void* p = ws + off;
    off += (bytes + 255) & ~(size_t)255;
    return p;
  };
  const size_t headsElems = (size_t)Bn * Hn * Sn * DKn;
  float*    qF    = (float*)alloc(headsElems * 4);
  float*    pF    = (float*)alloc(headsElems * 4);
  _Float16* qHh   = (_Float16*)alloc(headsElems * 2);
  _Float16* kHh   = (_Float16*)alloc(headsElems * 2);
  _Float16* vTh   = (_Float16*)alloc(headsElems * 2);
  _Float16* qnewH = (_Float16*)alloc((size_t)Mrows * Dn * 2);
  float*    oflat = (float*)alloc((size_t)Mrows * Dn * 4);
  _Float16* xH    = (_Float16*)alloc((size_t)Mrows * Dn * 2);
  _Float16* wT5   = (_Float16*)alloc((size_t)5 * Dn * Dn * 2);

  x_to_h<<<dim3((Mrows * Dn) / (256 * 8)), dim3(256), 0, stream>>>(x, xH);
  w_transpose<<<dim3(Dn / 32, Dn / 32, 5), dim3(256), 0, stream>>>(
      Wq, Wp, Wk, Wv, Wo, wT5);

  const size_t wStride = (size_t)Dn * Dn;
  dim3 gthr(256);
  dim3 ggrid(Dn / 128, Mrows / 128);
  const float qscale = 0.125f;  // 1/sqrt(DK) folded into q f16

  gemm128x128<<<ggrid, gthr, 0, stream>>>(xH, wT5 + 0 * wStride, qF, qHh,
                                          nullptr, nullptr, nullptr, nullptr,
                                          qscale);
  gemm128x128<<<ggrid, gthr, 0, stream>>>(xH, wT5 + 1 * wStride, pF, nullptr,
                                          nullptr, nullptr, nullptr, nullptr,
                                          1.0f);
  gemm128x128<<<ggrid, gthr, 0, stream>>>(xH, wT5 + 2 * wStride, nullptr, kHh,
                                          nullptr, nullptr, nullptr, nullptr,
                                          1.0f);
  gemm128x128<<<ggrid, gthr, 0, stream>>>(xH, wT5 + 3 * wStride, nullptr,
                                          nullptr, vTh, nullptr, nullptr,
                                          nullptr, 1.0f);

  attn_leapfrog<<<dim3(Bn * Hn * (Sn / 32)), dim3(128), 0, stream>>>(
      qHh, kHh, vTh, qF, pF, dtp, gatep, qnewH);

  gemm128x128<<<ggrid, gthr, 0, stream>>>(qnewH, wT5 + 4 * wStride, nullptr,
                                          nullptr, nullptr, oflat, bo, x, 1.0f);

  ln_rows<<<dim3(Mrows), dim3(256), 0, stream>>>(oflat, gamma, beta,
                                                 (float*)d_out);
}
